// CoCNModuleN_44349832299005
// MI455X (gfx1250) — compile-verified, matrix operand order checked by reading, not run on hardware
//
#include <hip/hip_runtime.h>
#include <math.h>

// ---------------------------------------------------------------------------
// CDNA5 (gfx1250) implementation of the CoCN reference network.
// Every dense contraction runs through v_wmma_f32_16x16x32_bf16 (wave32,
// bf16 operands, fp32 accumulate). All GEMM operands are stored K-contiguous
// (weights pre-transposed once) so fragments load as global_load_b128.
// Each wave computes a 32x32 tile = 2x2 WMMA subtiles for operand reuse.
// ---------------------------------------------------------------------------

typedef __attribute__((ext_vector_type(16))) __bf16 v16bf;
typedef __attribute__((ext_vector_type(8)))  float  v8f;
typedef unsigned int u32;
typedef __attribute__((ext_vector_type(4))) u32 u32x4;
typedef unsigned short bf16r;   // raw bf16 storage

union Frag {                    // 8 VGPRs = 16 bf16 per lane
    v16bf v;
    u32x4 q[2];
};

// float -> bf16 (round to nearest even)
__device__ __forceinline__ bf16r f2b(float f) {
    u32 u = __float_as_uint(f);
    if ((u & 0x7fffffffu) > 0x7f800000u) return (bf16r)((u >> 16) | 0x0040u); // qNaN
    u32 r = u + 0x7fffu + ((u >> 16) & 1u);
    return (bf16r)(r >> 16);
}

// ---------------------------------------------------------------------------
// Batched GEMM, both operands K-contiguous:
//   A: M x K, element (m,k) at m*lda + k          (lda % 8 == 0)
//   B: K x N, element (k,n) at n*ldb + k          (ldb % 8 == 0)
//   C: element (m,n) at batch + m*csr + n*csc  -> f32 C and/or bf16 Cb
// batch = (i0*d1 + i1)*d2 + i2, per-operand offset i0*s0 + i1*s1 + i2*s2.
// Requirements (hold for every call below): M % 64 == 0, K % 32 == 0,
// K/32 even. N is guarded.
// ---------------------------------------------------------------------------
struct GArg {
    const bf16r* A; const bf16r* B; float* C; bf16r* Cb; const float* bias;
    long as0, as1, as2, lda;
    long bs0, bs1, bs2, ldb;
    long cs0, cs1, cs2, csr, csc;
    int  d1, d2;
    int  M, N, K;
    int  relu, bias_mod;
};

__global__ __launch_bounds__(128) void k_gemm(GArg g) {
    const int lane = threadIdx.x & 31;
    const int wv   = threadIdx.x >> 5;
    const int row0 = blockIdx.y * 64 + (wv >> 1) * 32;
    const int col0 = blockIdx.x * 64 + (wv & 1) * 32;
    if (col0 >= g.N) return;                 // wave-uniform: EXEC stays all-1s

    int batch = blockIdx.z;
    int i2 = batch % g.d2; int rr = batch / g.d2;
    int i1 = rr % g.d1;    int i0 = rr / g.d1;
    const bf16r* Ab = g.A + i0 * g.as0 + i1 * g.as1 + i2 * g.as2;
    const bf16r* Bb = g.B + i0 * g.bs0 + i1 * g.bs1 + i2 * g.bs2;
    const long  coff = i0 * g.cs0 + i1 * g.cs1 + i2 * g.cs2;

    const int half = lane >> 4;
    const int l16  = lane & 15;

    // per-lane fragment base pointers (K-contiguous; half*8 folded in)
    const bf16r* pA0 = Ab + (long)(row0 + l16) * g.lda + half * 8;
    const bf16r* pA1 = pA0 + 16 * g.lda;
    const int gn0 = col0 + l16;
    const int gn1 = col0 + 16 + l16;
    const int cn0 = gn0 < g.N ? gn0 : g.N - 1;   // clamp OOB cols (stores masked)
    const int cn1 = gn1 < g.N ? gn1 : g.N - 1;
    const bf16r* pB0 = Bb + (long)cn0 * g.ldb + half * 8;
    const bf16r* pB1 = Bb + (long)cn1 * g.ldb + half * 8;

    v8f a00, a01, a10, a11;
#pragma unroll
    for (int r = 0; r < 8; ++r) { a00[r] = 0.f; a01[r] = 0.f; a10[r] = 0.f; a11[r] = 0.f; }

#pragma unroll 2
    for (int k0 = 0; k0 < g.K; k0 += 32) {
        Frag fa0, fa1, fb0, fb1;
        __builtin_prefetch(pA0 + k0 + 256, 0, 1);
        __builtin_prefetch(pB0 + k0 + 256, 0, 1);
        fa0.q[0] = *(const u32x4*)(pA0 + k0);
        fa0.q[1] = *(const u32x4*)(pA0 + k0 + 16);
        fa1.q[0] = *(const u32x4*)(pA1 + k0);
        fa1.q[1] = *(const u32x4*)(pA1 + k0 + 16);
        fb0.q[0] = *(const u32x4*)(pB0 + k0);
        fb0.q[1] = *(const u32x4*)(pB0 + k0 + 16);
        fb1.q[0] = *(const u32x4*)(pB1 + k0);
        fb1.q[1] = *(const u32x4*)(pB1 + k0 + 16);
        a00 = __builtin_amdgcn_wmma_f32_16x16x32_bf16(false, fa0.v, false, fb0.v, (short)0, a00, false, false);
        a01 = __builtin_amdgcn_wmma_f32_16x16x32_bf16(false, fa0.v, false, fb1.v, (short)0, a01, false, false);
        a10 = __builtin_amdgcn_wmma_f32_16x16x32_bf16(false, fa1.v, false, fb0.v, (short)0, a10, false, false);
        a11 = __builtin_amdgcn_wmma_f32_16x16x32_bf16(false, fa1.v, false, fb1.v, (short)0, a11, false, false);
    }

    // epilogue: C/D layout: lanes 0-15 rows M=r, lanes 16-31 rows M=8+r
#pragma unroll
    for (int rs = 0; rs < 2; ++rs) {
#pragma unroll
        for (int cs = 0; cs < 2; ++cs) {
            v8f acc = rs == 0 ? (cs == 0 ? a00 : a01) : (cs == 0 ? a10 : a11);
            const int gn = col0 + cs * 16 + l16;
            if (gn >= g.N) continue;
            const float bia = g.bias ? g.bias[g.bias_mod ? (gn % g.bias_mod) : gn] : 0.f;
            const int  gm0  = row0 + rs * 16 + half * 8;
            const long base = coff + (long)gm0 * g.csr + (long)gn * g.csc;
            if (g.Cb && g.csr == 1 && !g.C) {
                // rows contiguous: pack 8 bf16 into one b128 store
                u32 w[4];
#pragma unroll
                for (int e = 0; e < 4; ++e) {
                    float v0 = acc[2 * e] + bia, v1 = acc[2 * e + 1] + bia;
                    if (g.relu) { v0 = v0 > 0.f ? v0 : 0.f; v1 = v1 > 0.f ? v1 : 0.f; }
                    w[e] = (u32)f2b(v0) | ((u32)f2b(v1) << 16);
                }
                u32x4 pk; pk.x = w[0]; pk.y = w[1]; pk.z = w[2]; pk.w = w[3];
                *(u32x4*)(g.Cb + base) = pk;
            } else {
#pragma unroll
                for (int e = 0; e < 8; ++e) {
                    float v = acc[e] + bia;
                    if (g.relu) v = v > 0.f ? v : 0.f;
                    const long off = base + (long)e * g.csr;
                    if (g.C)  g.C[off]  = v;
                    if (g.Cb) g.Cb[off] = f2b(v);
                }
            }
        }
    }
}

// ------------------------------ helpers ------------------------------------

__global__ void k_cvt(const float* s, bf16r* d, long n) {
    for (long i = (long)blockIdx.x * blockDim.x + threadIdx.x; i < n;
         i += (long)gridDim.x * blockDim.x)
        d[i] = f2b(s[i]);
}

// batched transpose-convert: src (nb, R, C) f32 -> dst (nb, C, R) bf16
__global__ void k_cvtT(const float* s, bf16r* d, int R, int C, int nb) {
    long n = (long)nb * R * C;
    for (long i = (long)blockIdx.x * blockDim.x + threadIdx.x; i < n;
         i += (long)gridDim.x * blockDim.x) {
        int r = (int)(i % R); long t = i / R;
        int c = (int)(t % C); int b = (int)(t / C);
        d[i] = f2b(s[((long)b * R + r) * C + c]);
    }
}

// LayerNorm(+ReLU) over W=blockDim cols; trN==0: row-major out,
// trN==Nr: transposed out, row=(b,q): out[(b*W+t)*Nr + q]
__global__ void k_ln(const float* in, const float* gw, const float* bw,
                     bf16r* outb, int W, int trN) {
    __shared__ float sh[128];
    int row = blockIdx.x, t = threadIdx.x;
    float v = in[(long)row * W + t];
    sh[t] = v; __syncthreads();
    for (int o = 64; o > 0; o >>= 1) { if (t < o) sh[t] += sh[t + o]; __syncthreads(); }
    float mu = sh[0] / W; __syncthreads();
    float dv = v - mu;
    sh[t] = dv * dv; __syncthreads();
    for (int o = 64; o > 0; o >>= 1) { if (t < o) sh[t] += sh[t + o]; __syncthreads(); }
    float var = sh[0] / W;
    float y = dv * rsqrtf(var + 1e-5f) * gw[t] + bw[t];
    y = y > 0.f ? y : 0.f;
    long off = trN ? (((long)(row / trN) * W + t) * trN + (row % trN))
                   : ((long)row * W + t);
    outb[off] = f2b(y);
}

// g[bh,l,i,j] = b_e + sum_e W_e[e] * a_pad[bh,e, l*s+i, l*s+j]
__global__ void k_g(const float* a, const float* We, const float* be, float* g,
                    int BH, int E, int S, int L, int s, int F) {
    long n = (long)BH * L * F * F;
    for (long i = (long)blockIdx.x * blockDim.x + threadIdx.x; i < n;
         i += (long)gridDim.x * blockDim.x) {
        int j  = (int)(i % F); long r = i / F;
        int ii = (int)(r % F); r /= F;
        int l  = (int)(r % L); int bh = (int)(r / L);
        int row = l * s + ii, col = l * s + j;
        float acc = be[0];
        if (row < S && col < S)
            for (int e = 0; e < E; ++e)
                acc += We[e] * a[(((long)bh * E + e) * S + row) * S + col];
        g[i] = acc;
    }
}

// m[bh,l,i,dd] = (1/F) sum_j g[bh,l,i,j] * x_pad[bh, l*s+j, dd]  (-> bf16)
__global__ void k_m(const float* g, const float* x, bf16r* m,
                    int BH, int S, int L, int s, int F, int D) {
    long n = (long)BH * L * F * D;
    for (long i = (long)blockIdx.x * blockDim.x + threadIdx.x; i < n;
         i += (long)gridDim.x * blockDim.x) {
        int dd = (int)(i % D); long r = i / D;
        int ii = (int)(r % F); r /= F;
        int l  = (int)(r % L); int bh = (int)(r / L);
        const float* gp = g + (((long)bh * L + l) * F + ii) * F;
        float acc = 0.f;
        for (int j = 0; j < F; ++j) {
            int p = l * s + j;
            if (p < S) acc += gp[j] * x[((long)bh * S + p) * D + dd];
        }
        m[i] = f2b(acc / F);
    }
}

// x_new = res + mean_i x_old[l*s+i]   (writes f32 + bf16)
__global__ void k_xup(const float* res, const float* xo, float* xn, bf16r* xnb,
                      int BH, int S, int L, int s, int F, int D) {
    long n = (long)BH * L * D;
    for (long i = (long)blockIdx.x * blockDim.x + threadIdx.x; i < n;
         i += (long)gridDim.x * blockDim.x) {
        int dd = (int)(i % D); long r = i / D;
        int l  = (int)(r % L); int bh = (int)(r / L);
        float j = 0.f;
        for (int ii = 0; ii < F; ++ii) {
            int p = l * s + ii;
            if (p < S) j += xo[((long)bh * S + p) * D + dd];
        }
        float v = res[i] + j / F;
        xn[i] = v; xnb[i] = f2b(v);
    }
}

// a_next[be,l1,l2] = mean over F x F window (stride s) of a_pad
__global__ void k_pool(const float* a, float* an, int BHE, int S, int L, int s, int F) {
    long n = (long)BHE * L * L;
    for (long i = (long)blockIdx.x * blockDim.x + threadIdx.x; i < n;
         i += (long)gridDim.x * blockDim.x) {
        int l2 = (int)(i % L); long r = i / L;
        int l1 = (int)(r % L); int be = (int)(r / L);
        float acc = 0.f;
        for (int ii = 0; ii < F; ++ii) {
            int rrow = l1 * s + ii;
            if (rrow >= S) continue;
            const float* row = a + ((long)be * S + rrow) * S;
            for (int jj = 0; jj < F; ++jj) {
                int cc = l2 * s + jj;
                if (cc < S) acc += row[cc];
            }
        }
        an[i] = acc / (float)(F * F);
    }
}

// decoder unpool: x_new[n] = skip[n] + relu(gather-avg of c); f32 + bf16,
// plus optional transposed bf16 (BH, D, Lf) for the final perm^T GEMM.
__global__ void k_upd(const float* c, const float* skip, float* xn, bf16r* xnb,
                      bf16r* xtb, int BH, int Lc, int Lf, int s, int F, int D) {
    long n = (long)BH * Lf * D;
    for (long i = (long)blockIdx.x * blockDim.x + threadIdx.x; i < n;
         i += (long)gridDim.x * blockDim.x) {
        int dd = (int)(i % D); long r = i / D;
        int nn = (int)(r % Lf); int bh = (int)(r / Lf);
        float acc = 0.f; int cnt = 0;
        for (int ii = 0; ii < F; ++ii) {
            int t = nn - ii;
            if (t >= 0 && (t % s) == 0) {
                int l = t / s;
                if (l < Lc) { acc += c[(((long)bh * Lc + l) * F + ii) * D + dd]; ++cnt; }
            }
        }
        float up = cnt ? acc / (float)cnt : 0.f;
        up = up > 0.f ? up : 0.f;
        float v = skip[i] + up;
        xn[i] = v;
        xnb[i] = f2b(v);
        if (xtb) xtb[((long)bh * D + dd) * Lf + nn] = f2b(v);
    }
}

__global__ void k_logsoftmax(const float* in, float* out, int W) {
    __shared__ float sh[64];
    int row = blockIdx.x, t = threadIdx.x;
    float v = (t < W) ? in[(long)row * W + t] : -3.4e38f;
    sh[t] = v; __syncthreads();
    for (int o = 32; o > 0; o >>= 1) { if (t < o) sh[t] = fmaxf(sh[t], sh[t + o]); __syncthreads(); }
    float mx = sh[0]; __syncthreads();
    sh[t] = (t < W) ? __expf(v - mx) : 0.f; __syncthreads();
    for (int o = 32; o > 0; o >>= 1) { if (t < o) sh[t] += sh[t + o]; __syncthreads(); }
    float lse = logf(sh[0]);
    if (t < W) out[(long)row * W + t] = v - mx - lse;
}

// ------------------------------ host side ----------------------------------

static inline dim3 gsz(long n) {
    long b = (n + 255) / 256;
    if (b > 1048576) b = 1048576;
    if (b < 1) b = 1;
    return dim3((unsigned)b);
}

static void run_gemm(hipStream_t st, const GArg& g, int batches) {
    dim3 blk(128);
    dim3 grd((g.N + 63) / 64, (g.M + 63) / 64, batches);
    k_gemm<<<grd, blk, 0, st>>>(g);
}

extern "C" void kernel_launch(void* const* d_in, const int* in_sizes, int n_in,
                              void* d_out, int out_size, void* d_ws, size_t ws_size,
                              hipStream_t stream) {
    const int B = 2, H = 2, Nn = 1024, E = 2, DIN = 64, D = 128, NCLS = 40, F = 9, NL = 5;
    const int STR[5] = {1, 1, 2, 2, 1};
    const long NN = (long)Nn * Nn;

    const float* perm    = (const float*)d_in[0];
    const float* adj     = (const float*)d_in[1];
    const float* feats   = (const float*)d_in[2];
    const float* W_in    = (const float*)d_in[3];
    const float* b_in    = (const float*)d_in[4];
    const float* ln_in_g = (const float*)d_in[5];
    const float* ln_in_b = (const float*)d_in[6];
    const float* W_e     = (const float*)d_in[7];
    const float* b_e     = (const float*)d_in[8];
    const float* W_f     = (const float*)d_in[9];
    const float* b_f     = (const float*)d_in[10];
    const float* U       = (const float*)d_in[11];
    const float* b_u     = (const float*)d_in[12];
    const float* W_head  = (const float*)d_in[13];
    const float* b_head  = (const float*)d_in[14];
    const float* ln_o_g  = (const float*)d_in[15];
    const float* ln_o_b  = (const float*)d_in[16];
    const float* W_out   = (const float*)d_in[17];
    const float* b_out   = (const float*)d_in[18];

    // spatial / level sizes (mirrors the reference padding arithmetic)
    int S[6]; S[0] = Nn; int Lk[5];
    for (int k = 0; k < NL; ++k) {
        int s = STR[k];
        int bp = S[k] % s; if (bp == 0) bp = s;
        int below = F - bp;
        Lk[k] = (S[k] + below - F) / s + 1;
        S[k + 1] = Lk[k];
    }

    // workspace carving (256B aligned -> b128 fragment alignment holds)
    char* ws = (char*)d_ws; size_t cur = 0;
    auto alloc = [&](size_t bytes) -> void* {
        void* p = ws + cur; cur = (cur + bytes + 255) & ~(size_t)255; return p;
    };
    bf16r* permb  = (bf16r*)alloc((size_t)B * H * NN * 2);
    bf16r* permTb = (bf16r*)alloc((size_t)B * H * NN * 2);
    bf16r* adjb   = (bf16r*)alloc((size_t)B * E * NN * 2);
    bf16r* featb  = (bf16r*)alloc((size_t)B * Nn * DIN * 2);
    bf16r* WinT   = (bf16r*)alloc((size_t)DIN * D * 2);
    bf16r* WfT    = (bf16r*)alloc((size_t)NL * F * D * D * 2);
    bf16r* Upk    = (bf16r*)alloc((size_t)NL * F * D * D * 2);
    bf16r* WhT    = (bf16r*)alloc((size_t)H * D * D * 2);
    bf16r* WoT    = (bf16r*)alloc((size_t)D * NCLS * 2);
    float* xtmp   = (float*)alloc((size_t)B * Nn * D * 4);   // pre-LN rows (both LNs)
    bf16r* lnb    = (bf16r*)alloc((size_t)B * Nn * D * 2);   // x0^T / head LN out
    float* outs0  = (float*)alloc((size_t)B * H * Nn * D * 4);
    float* resb[5];
    for (int k = 0; k < NL; ++k) resb[k] = (float*)alloc((size_t)B * H * Lk[k] * D * 4);
    float* xA   = (float*)alloc((size_t)B * H * Nn * D * 4);
    bf16r* xAb  = (bf16r*)alloc((size_t)B * H * Nn * D * 2);
    float* xB   = (float*)alloc((size_t)B * H * Nn * D * 4);
    bf16r* xBb  = (bf16r*)alloc((size_t)B * H * Nn * D * 2);
    bf16r* xTf  = (bf16r*)alloc((size_t)B * H * D * Nn * 2); // final x transposed
    bf16r* tmpT = (bf16r*)alloc((size_t)B * H * E * NN * 2);
    float* aA   = (float*)alloc((size_t)B * H * E * NN * 4);
    float* aB   = (float*)alloc((size_t)B * H * E * NN * 4);
    float* gbuf = (float*)alloc((size_t)B * H * Nn * F * F * 4);
    bf16r* mb   = (bf16r*)alloc((size_t)B * H * Nn * F * D * 2);
    float* cbuf = (float*)alloc((size_t)B * H * Nn * F * D * 4);
    bf16r* hpack = (bf16r*)alloc((size_t)B * Nn * H * D * 2);
    float* logits = (float*)alloc((size_t)B * Nn * NCLS * 4);
    if (cur > ws_size) return;

    // --- conversions: bf16 copies + K-contiguous (transposed) weights ---
    k_cvt<<<gsz((long)B * H * NN), 256, 0, stream>>>(perm, permb, (long)B * H * NN);
    k_cvt<<<gsz((long)B * E * NN), 256, 0, stream>>>(adj, adjb, (long)B * E * NN);
    k_cvt<<<gsz((long)B * Nn * DIN), 256, 0, stream>>>(feats, featb, (long)B * Nn * DIN);
    k_cvtT<<<gsz((long)B * H * NN), 256, 0, stream>>>(perm, permTb, Nn, Nn, B * H);
    k_cvtT<<<gsz((long)DIN * D), 256, 0, stream>>>(W_in, WinT, DIN, D, 1);
    k_cvtT<<<gsz((long)NL * F * D * D), 256, 0, stream>>>(W_f, WfT, F * D, D, NL);
    k_cvtT<<<gsz((long)NL * F * D * D), 256, 0, stream>>>(U, Upk, D, D, NL * F);
    k_cvtT<<<gsz((long)H * D * D), 256, 0, stream>>>(W_head, WhT, H * D, D, 1);
    k_cvtT<<<gsz((long)D * NCLS), 256, 0, stream>>>(W_out, WoT, D, NCLS, 1);

    GArg g;

    // 1) x = features @ W_in + b_in     (2048 x 64 x 128)
    g = GArg{}; g.A = featb; g.B = WinT; g.C = xtmp; g.bias = b_in;
    g.lda = DIN; g.ldb = DIN; g.csr = D; g.csc = 1;
    g.d1 = 1; g.d2 = 1; g.M = B * Nn; g.N = D; g.K = DIN;
    run_gemm(stream, g, 1);

    // 2) relu(LN(x)) -> x0^T bf16 (per b: D x Nn)
    k_ln<<<B * Nn, D, 0, stream>>>(xtmp, ln_in_g, ln_in_b, lnb, D, Nn);

    // 3) x = perm @ x0   per (b,h)   -> outs0 (f32)
    g = GArg{}; g.A = permb; g.B = lnb; g.C = outs0;
    g.as0 = (long)H * NN; g.as1 = NN; g.lda = Nn;
    g.bs0 = (long)D * Nn; g.ldb = Nn;
    g.cs0 = (long)H * Nn * D; g.cs1 = (long)Nn * D; g.csr = D; g.csc = 1;
    g.d1 = H; g.d2 = 1; g.M = Nn; g.N = D; g.K = Nn;
    run_gemm(stream, g, B * H);

    // 4) tmp[b,h,e,p,k] = sum_q adj[b,e,p,q] perm[b,h,k,q] -> bf16 TRANSPOSED
    g = GArg{}; g.A = adjb; g.B = permb; g.Cb = tmpT;
    g.as0 = (long)E * NN; g.as2 = NN; g.lda = Nn;
    g.bs0 = (long)H * NN; g.bs1 = NN; g.ldb = Nn;
    g.cs0 = (long)H * E * NN; g.cs1 = (long)E * NN; g.cs2 = NN;
    g.csr = 1; g.csc = Nn;                       // store tmp^T: K-contig for step 5
    g.d1 = H; g.d2 = E; g.M = Nn; g.N = Nn; g.K = Nn;
    run_gemm(stream, g, B * H * E);

    // 5) a[b,h,e,m,k] = sum_p perm[b,h,m,p] tmp[b,h,e,p,k]   -> f32
    g = GArg{}; g.A = permb; g.B = tmpT; g.C = aA;
    g.as0 = (long)H * NN; g.as1 = NN; g.lda = Nn;
    g.bs0 = (long)H * E * NN; g.bs1 = (long)E * NN; g.bs2 = NN; g.ldb = Nn;
    g.cs0 = (long)H * E * NN; g.cs1 = (long)E * NN; g.cs2 = NN; g.csr = Nn; g.csc = 1;
    g.d1 = H; g.d2 = E; g.M = Nn; g.N = Nn; g.K = Nn;
    run_gemm(stream, g, B * H * E);

    // --- encoder ---
    const int BH = B * H;
    float* xcur = outs0; float* xnxt = xA; bf16r* xnxtb = xAb;
    float* acur = aA;    float* anxt = aB;
    for (int k = 0; k < NL; ++k) {
        int s = STR[k], Sk = S[k], L = Lk[k];
        k_g<<<gsz((long)BH * L * F * F), 256, 0, stream>>>(
            acur, W_e + (long)k * E, b_e + k, gbuf, BH, E, Sk, L, s, F);
        k_m<<<gsz((long)BH * L * F * D), 256, 0, stream>>>(
            gbuf, xcur, mb, BH, Sk, L, s, F, D);
        // res = relu(m @ W_f[k] + b_f[k])   (BH*L x 1152 x 128)
        g = GArg{}; g.A = mb; g.B = WfT + (long)k * F * D * D; g.C = resb[k];
        g.bias = b_f + (long)k * D; g.relu = 1;
        g.lda = (long)F * D; g.ldb = (long)F * D; g.csr = D; g.csc = 1;
        g.d1 = 1; g.d2 = 1; g.M = BH * L; g.N = D; g.K = F * D;
        run_gemm(stream, g, 1);
        k_xup<<<gsz((long)BH * L * D), 256, 0, stream>>>(
            resb[k], xcur, xnxt, xnxtb, BH, Sk, L, s, F, D);
        if (k < NL - 1) {
            k_pool<<<gsz((long)BH * E * L * L), 256, 0, stream>>>(
                acur, anxt, BH * E, Sk, L, s, F);
            float* t = acur; acur = anxt; anxt = t;
        }
        xcur = xnxt;
        if (xnxt == xA) { xnxt = xB; xnxtb = xBb; } else { xnxt = xA; xnxtb = xAb; }
    }
    bf16r* xcurb = (xcur == xA) ? xAb : xBb;

    // --- decoder ---
    for (int k = NL - 1; k >= 0; --k) {
        int s = STR[k];
        int Lc = Lk[k];
        int Lf = (k > 0) ? Lk[k - 1] : Nn;
        // c = x @ U[k] + b_u[k]   (BH*Lc x 128 x 1152)
        g = GArg{}; g.A = xcurb; g.B = Upk + (long)k * F * D * D; g.C = cbuf;
        g.bias = b_u + (long)k * D; g.bias_mod = D;
        g.lda = D; g.ldb = D; g.csr = (long)F * D; g.csc = 1;
        g.d1 = 1; g.d2 = 1; g.M = BH * Lc; g.N = F * D; g.K = D;
        run_gemm(stream, g, 1);
        const float* skip = (k == 0) ? outs0 : resb[k - 1];
        k_upd<<<gsz((long)BH * Lf * D), 256, 0, stream>>>(
            cbuf, skip, xnxt, xnxtb, (k == 0) ? xTf : (bf16r*)nullptr,
            BH, Lc, Lf, s, F, D);
        xcur = xnxt; xcurb = xnxtb;
        if (xnxt == xA) { xnxt = xB; xnxtb = xBb; } else { xnxt = xA; xnxtb = xAb; }
    }

    // 6) out = perm^T @ x per (b,h) -> bf16 directly in (B,N,h*D) layout
    g = GArg{}; g.A = permTb; g.B = xTf; g.Cb = hpack;
    g.as0 = (long)H * NN; g.as1 = NN; g.lda = Nn;
    g.bs0 = (long)H * D * Nn; g.bs1 = (long)D * Nn; g.ldb = Nn;
    g.cs0 = (long)Nn * H * D; g.cs1 = D; g.csr = (long)H * D; g.csc = 1;
    g.d1 = H; g.d2 = 1; g.M = Nn; g.N = D; g.K = Nn;
    run_gemm(stream, g, B * H);

    // 7) head: (B*N x 256) @ (256 x 128) + b_head
    g = GArg{}; g.A = hpack; g.B = WhT; g.C = xtmp; g.bias = b_head;
    g.lda = (long)H * D; g.ldb = (long)H * D; g.csr = D; g.csc = 1;
    g.d1 = 1; g.d2 = 1; g.M = B * Nn; g.N = D; g.K = H * D;
    run_gemm(stream, g, 1);

    // 8) relu(LN(head)) -> bf16 row-major
    k_ln<<<B * Nn, D, 0, stream>>>(xtmp, ln_o_g, ln_o_b, lnb, D, 0);

    // 9) logits = hb @ W_out + b_out   (B*N x 128 x 40)
    g = GArg{}; g.A = lnb; g.B = WoT; g.C = logits; g.bias = b_out;
    g.lda = D; g.ldb = D; g.csr = NCLS; g.csc = 1;
    g.d1 = 1; g.d2 = 1; g.M = B * Nn; g.N = NCLS; g.K = D;
    run_gemm(stream, g, 1);

    // 10) log_softmax -> d_out
    k_logsoftmax<<<B * Nn, 64, 0, stream>>>(logits, (float*)d_out, NCLS);
}
